// SwinBlock_57836029608160
// MI455X (gfx1250) — matrix operand order repaired and checked by reading, not compiled, not run on hardware
//
#include <hip/hip_runtime.h>
#include <hip/hip_bf16.h>

typedef __attribute__((ext_vector_type(16))) __bf16 bf16x16;
typedef __attribute__((ext_vector_type(8)))  float  f32x8;
typedef __attribute__((ext_vector_type(4)))  unsigned int u32x4;
typedef __attribute__((ext_vector_type(8)))  int    i32x8;
typedef __attribute__((ext_vector_type(4)))  int    i32x4;

// ---------------------------------------------------------------------------
// WMMA fragment loaders (wave32); A/B use this code's contiguous k order, not the ISA table order:
//   A (16xK, 16-bit): lane -> M=lane&15; K base = (lane<16)?0:16, 16 sequential K values
//   B (Kx16, 16-bit): lane -> N=lane&15; K base = (lane<16)?0:16, 16 sequential K values
//   C/D (16x16 f32):  lane -> N=lane&15; vgpr r -> M = r + 8*(lane>=16)
// ---------------------------------------------------------------------------
__device__ __forceinline__ bf16x16 load_a_frag(const __bf16* src, int ld) {
  const int lane = threadIdx.x & 31;
  const __bf16* row = src + (size_t)(lane & 15) * ld;
  const int base = (lane < 16) ? 0 : 16;
  bf16x16 a;
#pragma unroll
  for (int j = 0; j < 8; ++j) {
    const int k = base + 2 * j;
    a[2 * j]     = row[k];
    a[2 * j + 1] = row[k + 1];
  }
  return a;
}

// B fragment from an N-major (transposed) tile: src[n*ld + k]
__device__ __forceinline__ bf16x16 load_b_frag_nk(const __bf16* src, int ld) {
  const int lane = threadIdx.x & 31;
  const __bf16* p = src + (size_t)(lane & 15) * ld + ((lane < 16) ? 0 : 16);
  bf16x16 b;
#pragma unroll
  for (int e = 0; e < 16; ++e) b[e] = p[e];
  return b;
}

// B fragment from a K-major tile: src[k*ld + n]
__device__ __forceinline__ bf16x16 load_b_frag_kn(const __bf16* src, int ld) {
  const int lane = threadIdx.x & 31;
  const int n  = lane & 15;
  const int kb = (lane < 16) ? 0 : 16;
  bf16x16 b;
#pragma unroll
  for (int e = 0; e < 16; ++e) b[e] = src[(size_t)(kb + e) * ld + n];
  return b;
}

__device__ __forceinline__ f32x8 wmma_bf16(bf16x16 a, bf16x16 b, f32x8 c) {
  return __builtin_amdgcn_wmma_f32_16x16x32_bf16(false, a, false, b, (short)0, c,
                                                 false, false);
}

// ---------------------------------------------------------------------------
// Tensor Data Mover: DMA a 2-D tile (rows x 16 dwords) from global into LDS,
// with HW padding of 4 dwords every 16 dwords -> padded LDS rows of 40 bf16.
// D# layout per CDNA5 ISA 8.3-8.5. Tracked by TENSORcnt.
// ---------------------------------------------------------------------------
__device__ __forceinline__ void tdm_load_tile_2d(unsigned lds_off, const void* gptr,
                                                 unsigned rows, unsigned stride_dw) {
  const unsigned long long ga = (unsigned long long)(uintptr_t)gptr;
  u32x4 g0;
  g0[0] = 1u;                                    // count=1 (valid user descriptor)
  g0[1] = lds_off;                               // lds_addr (bytes)
  g0[2] = (unsigned)ga;                          // global_addr[31:0]  (bits 95:64)
  g0[3] = (unsigned)((ga >> 32) & 0x01ffffffu)   // global_addr[56:32]
        | (2u << 30);                            // type = 2 ("image")
  i32x8 g1;
  g1[0] = (int)((2u << 16)       // data_size = 4 bytes
              | (1u << 20)       // pad_enable
              | (3u << 22)       // pad_interval: 16 dwords
              | (3u << 25));     // pad_amount:   4 dwords
  g1[1] = (int)((stride_dw & 0xffffu) << 16);                   // tensor_dim0[15:0]
  g1[2] = (int)((stride_dw >> 16) | ((rows & 0xffffu) << 16));  // td0 hi / td1 lo
  g1[3] = (int)((rows >> 16) | (16u << 16));                    // td1 hi / tile_dim0=16dw
  g1[4] = (int)(rows & 0xffffu);                                // tile_dim1 (tile_dim2=0)
  g1[5] = (int)stride_dw;                                       // tensor_dim0_stride lo32
  g1[6] = 0;                                                    // stride hi / td1_stride
  g1[7] = 0;
  const i32x4 gz = {0, 0, 0, 0};
#if defined(__clang_major__) && (__clang_major__ >= 23)
  const i32x8 gz8 = {0, 0, 0, 0, 0, 0, 0, 0};
  __builtin_amdgcn_tensor_load_to_lds(g0, g1, gz, gz, gz8, 0);
#else
  __builtin_amdgcn_tensor_load_to_lds(g0, g1, gz, gz, 0);
#endif
}

// ---------------------------------------------------------------------------
// fp32 -> bf16 weight conversion WITH transpose: out[n*K + k] = in[k*Nd + n].
// (Transposed weights make the GEMM B tile a row-major 2-D slice -> TDM-able,
// and give contiguous per-lane B-fragment reads from LDS.)
// ---------------------------------------------------------------------------
__global__ void f2bf_t_kernel(const float* __restrict__ in, __bf16* __restrict__ out,
                              int K, int Nd) {
  const int i = blockIdx.x * 256 + threadIdx.x;
  if (i < K * Nd) {
    const int k = i / Nd, n = i % Nd;
    out[(size_t)n * K + k] = (__bf16)in[i];
  }
}

// ---------------------------------------------------------------------------
// LayerNorm over C=128, one wave (32 lanes * 4ch) per row; optional window
// partition mapping on the output row index (win=(h%8,w%8), token=(h/8,w/8)).
// ---------------------------------------------------------------------------
__global__ __launch_bounds__(256)
void ln_kernel(const float* __restrict__ x, const float* __restrict__ g,
               const float* __restrict__ b, __bf16* __restrict__ out, int partition) {
  const int lane = threadIdx.x & 31;
  const int wave = threadIdx.x >> 5;
  const size_t row = (size_t)blockIdx.x * 8 + wave;   // 0 .. 131071
  const float* xr = x + row * 128;
  const float4 v = ((const float4*)xr)[lane];
  float vv[4] = {v.x, v.y, v.z, v.w};
  float s  = vv[0] + vv[1] + vv[2] + vv[3];
  float sq = vv[0] * vv[0] + vv[1] * vv[1] + vv[2] * vv[2] + vv[3] * vv[3];
#pragma unroll
  for (int m = 1; m < 32; m <<= 1) {
    s  += __shfl_xor(s, m, 32);
    sq += __shfl_xor(sq, m, 32);
  }
  const float mean = s * (1.0f / 128.0f);
  const float var  = sq * (1.0f / 128.0f) - mean * mean;
  const float rstd = rsqrtf(var + 1e-5f);
  size_t orow = row;
  if (partition) {
    const int bimg = (int)(row >> 12);
    const int p = (int)row & 4095;
    const int h = p >> 6, w = p & 63;
    const int win   = bimg * 64 + (h & 7) * 8 + (w & 7);
    const int token = (h >> 3) * 8 + (w >> 3);
    orow = (size_t)win * 64 + token;
  }
  const int c = lane * 4;
  __align__(8) __bf16 o[4];
#pragma unroll
  for (int t = 0; t < 4; ++t)
    o[t] = (__bf16)((vv[t] - mean) * rstd * g[c + t] + b[c + t]);
  *(uint2*)&out[orow * 128 + c] = *(const uint2*)o;
}

// ---------------------------------------------------------------------------
// Generic bf16 GEMM: C[M x Nd] = A[M x K] * Wt^T + bias, Wt is [Nd x K], M=131072.
// Block: 256 thr (8 waves), tile 64M x 128N; wave tile 32x32 (2x2 WMMA).
// A and B tiles DMA'd into double-buffered LDS by the Tensor Data Mover
// (wave 0 issues, TENSORcnt-pipelined), HW-padded to ld=40 bf16.
// Epilogue modes: 0 = store bf16 (QKV), 1 = proj + un-partition + residual,
//                 2 = exact GELU -> bf16 (MLP1), 3 = +bias +msa -> f32 (MLP2)
// ---------------------------------------------------------------------------
__global__ __launch_bounds__(256)
void gemm_bf16_kernel(const __bf16* __restrict__ A, const __bf16* __restrict__ Wt,
                      const float* __restrict__ bias, int K, int Nd, int mode,
                      void* __restrict__ out_v, const float* __restrict__ res) {
  __shared__ __align__(16) __bf16 As[2][64][40];
  __shared__ __align__(16) __bf16 Bs[2][128][40];

  const int tid = threadIdx.x;
  const int m0 = blockIdx.x * 64;
  const int n0 = blockIdx.y * 128;
  const int wave = tid >> 5;
  const int mb = (wave & 1) * 32;
  const int nb = (wave >> 1) * 32;
  const unsigned stride_dw = (unsigned)K >> 1;   // row stride in dwords

  f32x8 acc[2][2];
#pragma unroll
  for (int i = 0; i < 2; ++i)
#pragma unroll
    for (int j = 0; j < 2; ++j) acc[i][j] = {};

  if (wave == 0) {  // prologue: DMA first tiles into buffer 0
    tdm_load_tile_2d((unsigned)(uintptr_t)&As[0][0][0],
                     A + (size_t)m0 * K, 64, stride_dw);
    tdm_load_tile_2d((unsigned)(uintptr_t)&Bs[0][0][0],
                     Wt + (size_t)n0 * K, 128, stride_dw);
  }

  int cur = 0;
  for (int kk = 0; kk < K; kk += 32, cur ^= 1) {
    if (wave == 0) {
      if (kk + 32 < K) {   // prefetch next K-slice into the other buffer
        tdm_load_tile_2d((unsigned)(uintptr_t)&As[cur ^ 1][0][0],
                         A + (size_t)m0 * K + kk + 32, 64, stride_dw);
        tdm_load_tile_2d((unsigned)(uintptr_t)&Bs[cur ^ 1][0][0],
                         Wt + (size_t)n0 * K + kk + 32, 128, stride_dw);
        __builtin_amdgcn_s_wait_tensorcnt(2);  // current tile done (in-order)
      } else {
        __builtin_amdgcn_s_wait_tensorcnt(0);
      }
    }
    __syncthreads();  // current buffer ready for all waves

    bf16x16 af[2], bfr[2];
#pragma unroll
    for (int i = 0; i < 2; ++i) af[i] = load_a_frag(&As[cur][mb + i * 16][0], 40);
#pragma unroll
    for (int j = 0; j < 2; ++j) bfr[j] = load_b_frag_nk(&Bs[cur][nb + j * 16][0], 40);
#pragma unroll
    for (int i = 0; i < 2; ++i)
#pragma unroll
      for (int j = 0; j < 2; ++j) acc[i][j] = wmma_bf16(af[i], bfr[j], acc[i][j]);
    __syncthreads();  // all waves done reading before this buffer is re-filled
  }

  const int lane = tid & 31;
  const int nl = lane & 15;
  const int mo = (lane < 16) ? 0 : 8;
#pragma unroll
  for (int i = 0; i < 2; ++i) {
#pragma unroll
    for (int j = 0; j < 2; ++j) {
      const int gnc = n0 + nb + j * 16 + nl;
      const float bv = bias[gnc];
#pragma unroll
      for (int r = 0; r < 8; ++r) {
        const int gm = m0 + mb + i * 16 + mo + r;
        const float v = acc[i][j][r] + bv;
        if (mode == 0) {
          ((__bf16*)out_v)[(size_t)gm * Nd + gnc] = (__bf16)v;
        } else if (mode == 2) {
          const float gl = 0.5f * v * (1.0f + erff(v * 0.70710678118654752f));
          ((__bf16*)out_v)[(size_t)gm * Nd + gnc] = (__bf16)gl;
        } else if (mode == 1) {
          const int win = gm >> 6, token = gm & 63;
          const int bimg = win >> 6, wl = win & 63;
          const int h = ((token >> 3) << 3) | (wl >> 3);
          const int w = ((token & 7) << 3) | (wl & 7);
          const size_t dst = (((size_t)bimg * 4096) + h * 64 + w) * 128 + gnc;
          ((float*)out_v)[dst] = res[dst] + v;       // msa = x + rev(proj)
        } else {
          const size_t dst = (size_t)gm * 128 + gnc;
          ((float*)out_v)[dst] = res[dst] + v;       // out = msa + mlp
        }
      }
    }
  }
}

// ---------------------------------------------------------------------------
// Fused windowed attention: block = 1 window (64 tokens), wave = 1 head.
// S = (Q K^T)/sqrt(32) + rel_bias; softmax over keys; O = P V.
// HD=32 -> energy needs one 16x16x32 WMMA K-step; PV needs two.
// ---------------------------------------------------------------------------
__global__ __launch_bounds__(128)
void attn_kernel(const __bf16* __restrict__ qkv, const float* __restrict__ table,
                 __bf16* __restrict__ attn_out) {
  __shared__ float biasS[225 * 4];
  __shared__ __align__(16) __bf16 Ps[4][64][72];   // padded ld to spread LDS banks

  const int tid = threadIdx.x;
  const int head = tid >> 5;
  const int lane = tid & 31;
  const int win = blockIdx.x;

  for (int i = tid; i < 900; i += 128) biasS[i] = table[i];
  __syncthreads();

  const __bf16* base = qkv + (size_t)win * 64 * 384;
  const __bf16* Qp = base + head * 96;
  const __bf16* Kp = base + head * 96 + 32;
  const __bf16* Vp = base + head * 96 + 64;

  bf16x16 qa[4];
#pragma unroll
  for (int mt = 0; mt < 4; ++mt) qa[mt] = load_a_frag(Qp + (size_t)mt * 16 * 384, 384);

  f32x8 s[4][4];
#pragma unroll
  for (int nt = 0; nt < 4; ++nt) {
    const bf16x16 kb = load_b_frag_nk(Kp + (size_t)nt * 16 * 384, 384);
#pragma unroll
    for (int mt = 0; mt < 4; ++mt) {
      f32x8 z = {};
      s[mt][nt] = wmma_bf16(qa[mt], kb, z);
    }
  }

  const int nl = lane & 15;
  const int mo = (lane < 16) ? 0 : 8;
  const float scale = 0.17677669529663687f;  // 32^-0.5

#pragma unroll
  for (int mt = 0; mt < 4; ++mt) {
#pragma unroll
    for (int r = 0; r < 8; ++r) {
      const int q = mt * 16 + mo + r;
      float v[4];
#pragma unroll
      for (int nt = 0; nt < 4; ++nt) {
        const int k = nt * 16 + nl;
        const int idx = ((q >> 3) - (k >> 3) + 7) * 15 + ((q & 7) - (k & 7) + 7);
        v[nt] = s[mt][nt][r] * scale + biasS[idx * 4 + head];
      }
      float mx = fmaxf(fmaxf(v[0], v[1]), fmaxf(v[2], v[3]));
#pragma unroll
      for (int m = 1; m < 16; m <<= 1) mx = fmaxf(mx, __shfl_xor(mx, m, 32));
      float sum = 0.0f;
#pragma unroll
      for (int nt = 0; nt < 4; ++nt) { v[nt] = __expf(v[nt] - mx); sum += v[nt]; }
#pragma unroll
      for (int m = 1; m < 16; m <<= 1) sum += __shfl_xor(sum, m, 32);
      const float inv = 1.0f / sum;
#pragma unroll
      for (int nt = 0; nt < 4; ++nt)
        Ps[head][q][nt * 16 + nl] = (__bf16)(v[nt] * inv);
    }
  }
  __syncthreads();

  f32x8 o[4][2];
#pragma unroll
  for (int mt = 0; mt < 4; ++mt)
#pragma unroll
    for (int nt = 0; nt < 2; ++nt) o[mt][nt] = {};

#pragma unroll
  for (int ks = 0; ks < 2; ++ks) {
    bf16x16 pa[4];
#pragma unroll
    for (int mt = 0; mt < 4; ++mt)
      pa[mt] = load_a_frag(&Ps[head][mt * 16][ks * 32], 72);
#pragma unroll
    for (int nt = 0; nt < 2; ++nt) {
      const bf16x16 vb = load_b_frag_kn(Vp + (size_t)ks * 32 * 384 + nt * 16, 384);
#pragma unroll
      for (int mt = 0; mt < 4; ++mt) o[mt][nt] = wmma_bf16(pa[mt], vb, o[mt][nt]);
    }
  }

#pragma unroll
  for (int mt = 0; mt < 4; ++mt)
#pragma unroll
    for (int nt = 0; nt < 2; ++nt)
#pragma unroll
      for (int r = 0; r < 8; ++r) {
        const int q = mt * 16 + mo + r;
        attn_out[((size_t)win * 64 + q) * 128 + head * 32 + nt * 16 + nl] =
            (__bf16)o[mt][nt][r];
      }
}

// ---------------------------------------------------------------------------
extern "C" void kernel_launch(void* const* d_in, const int* in_sizes, int n_in,
                              void* d_out, int out_size, void* d_ws, size_t ws_size,
                              hipStream_t stream) {
  (void)in_sizes; (void)n_in; (void)out_size; (void)ws_size;
  const float* x      = (const float*)d_in[0];
  const float* qkv_w  = (const float*)d_in[1];
  const float* qkv_b  = (const float*)d_in[2];
  const float* proj_w = (const float*)d_in[3];
  const float* proj_b = (const float*)d_in[4];
  const float* rel    = (const float*)d_in[5];
  const float* ln1_g  = (const float*)d_in[6];
  const float* ln1_b  = (const float*)d_in[7];
  const float* ln2_g  = (const float*)d_in[8];
  const float* ln2_b  = (const float*)d_in[9];
  const float* mlp_w1 = (const float*)d_in[10];
  const float* mlp_b1 = (const float*)d_in[11];
  const float* mlp_w2 = (const float*)d_in[12];
  const float* mlp_b2 = (const float*)d_in[13];

  char* ws = (char*)d_ws;
  __bf16* xp     = (__bf16*)(ws);                     // 33,554,432 B (LN1 partitioned)
  __bf16* qkvb   = (__bf16*)(ws + 33554432ull);       // 100,663,296 B
  __bf16* hidden = (__bf16*)(ws);                     // 134,217,728 B (aliases xp+qkv, both dead)
  __bf16* attn   = (__bf16*)(ws + 134217728ull);      // 33,554,432 B
  float*  msa    = (float*)(ws + 167772160ull);       // 67,108,864 B
  __bf16* h2     = (__bf16*)(ws + 234881024ull);      // 33,554,432 B
  __bf16* wqkv   = (__bf16*)(ws + 268435456ull);      // 98,304 B   [384 x 128] (transposed)
  __bf16* wproj  = (__bf16*)(ws + 268533760ull);      // 32,768 B   [128 x 128] (transposed)
  __bf16* w1     = (__bf16*)(ws + 268566528ull);      // 131,072 B  [512 x 128] (transposed)
  __bf16* w2     = (__bf16*)(ws + 268697600ull);      // 131,072 B  [128 x 512] (transposed)

  // one-time weight conversion to bf16 + transpose (Wt[n][k])
  f2bf_t_kernel<<<(49152 + 255) / 256, 256, 0, stream>>>(qkv_w, wqkv, 128, 384);
  f2bf_t_kernel<<<(16384 + 255) / 256, 256, 0, stream>>>(proj_w, wproj, 128, 128);
  f2bf_t_kernel<<<(65536 + 255) / 256, 256, 0, stream>>>(mlp_w1, w1, 128, 512);
  f2bf_t_kernel<<<(65536 + 255) / 256, 256, 0, stream>>>(mlp_w2, w2, 512, 128);

  // LN1 + window partition
  ln_kernel<<<16384, 256, 0, stream>>>(x, ln1_g, ln1_b, xp, 1);
  // QKV projection: [131072,128] x [128,384]
  gemm_bf16_kernel<<<dim3(2048, 3), 256, 0, stream>>>(xp, wqkv, qkv_b, 128, 384, 0,
                                                      qkvb, nullptr);
  // windowed multi-head attention
  attn_kernel<<<2048, 128, 0, stream>>>(qkvb, rel, attn);
  // output projection + reverse partition + residual -> msa (fp32)
  gemm_bf16_kernel<<<dim3(2048, 1), 256, 0, stream>>>(attn, wproj, proj_b, 128, 128, 1,
                                                      msa, x);
  // LN2 (identity layout)
  ln_kernel<<<16384, 256, 0, stream>>>(msa, ln2_g, ln2_b, h2, 0);
  // MLP fc1 + exact GELU: [131072,128] x [128,512]
  gemm_bf16_kernel<<<dim3(2048, 4), 256, 0, stream>>>(h2, w1, mlp_b1, 128, 512, 2,
                                                      hidden, nullptr);
  // MLP fc2 + residual -> out (fp32): [131072,512] x [512,128]
  gemm_bf16_kernel<<<dim3(2048, 1), 256, 0, stream>>>(hidden, w2, mlp_b2, 512, 128, 3,
                                                      (float*)d_out, msa);
}
